// MdpTorchKernel_60318520705714
// MI455X (gfx1250) — matrix operand order repaired and checked by reading, not compiled, not run on hardware
//
#include <hip/hip_runtime.h>
#include <hip/hip_bf16.h>

// ---------------------------------------------------------------------------
// MDP backward scan, MI455X (gfx1250).
// Dominant op per step: Phi = T4(131072x512) @ [Vr,Vc].  Memory bound.
//   fp32: 256 steps * 256MB = 64 GB HBM  -> >= 2.75 ms @ 23.3 TB/s
//   bf16: T4 converted once to 128MB -> fits 192MB L2; all 256 passes are
//         L2-resident.  WMMA f32_16x16x32_bf16 with fp32 accumulation.
// B operand (Vr/Vc, 2KB) is TDM-staged into LDS once per block and read with
// ds_load_b128 in the hot loop; zero columns come from a zeroed LDS row so the
// loop is branchless (no EXEC save/restore per iteration).
// ---------------------------------------------------------------------------

#define LEN_H 256
#define N_M   512
#define N_AX  16
#define N_AY  16
#define N_G   64

static constexpr size_t T4_ELEMS = (size_t)N_M * N_AX * N_AY * N_M; // 67,108,864

typedef __attribute__((ext_vector_type(16))) __bf16       v16bf;
typedef __attribute__((ext_vector_type(8)))  float        v8f;
typedef __attribute__((ext_vector_type(4)))  unsigned int u32x4;
typedef __attribute__((ext_vector_type(4)))  float        f32x4;
typedef __attribute__((ext_vector_type(4)))  unsigned int v4u;
typedef __attribute__((ext_vector_type(8)))  int          v8i;
typedef __attribute__((ext_vector_type(4)))  int          v4i;

__device__ __forceinline__ unsigned short f32_to_bf16_rne(float f) {
  unsigned u = __float_as_uint(f);
  u += 0x7FFFu + ((u >> 16) & 1u);      // round-to-nearest-even
  return (unsigned short)(u >> 16);
}

// --- one-time: T4 fp32 -> bf16 (128MB, L2-resident thereafter) -------------
__global__ __launch_bounds__(256) void k_convert_t4(
    const float* __restrict__ t4, unsigned short* __restrict__ t4bf) {
  const size_t total8 = T4_ELEMS / 8;
  const size_t stride = (size_t)gridDim.x * blockDim.x;
  for (size_t i = (size_t)blockIdx.x * blockDim.x + threadIdx.x; i < total8; i += stride) {
    const f32x4* src = reinterpret_cast<const f32x4*>(t4 + i * 8);
    f32x4 a = src[0], b = src[1];
    u32x4 o;
    o.x = (unsigned)f32_to_bf16_rne(a.x) | ((unsigned)f32_to_bf16_rne(a.y) << 16);
    o.y = (unsigned)f32_to_bf16_rne(a.z) | ((unsigned)f32_to_bf16_rne(a.w) << 16);
    o.z = (unsigned)f32_to_bf16_rne(b.x) | ((unsigned)f32_to_bf16_rne(b.y) << 16);
    o.w = (unsigned)f32_to_bf16_rne(b.z) | ((unsigned)f32_to_bf16_rne(b.w) << 16);
    *reinterpret_cast<u32x4*>(t4bf + i * 8) = o;
  }
}

// --- prep: px softmax over x, per t ----------------------------------------
__global__ void k_px(const float* __restrict__ pv, float* __restrict__ px) {
  int t = blockIdx.x * blockDim.x + threadIdx.x;
  if (t >= LEN_H) return;
  const float* r = pv + t * N_AX;
  float m = r[0];
  for (int i = 1; i < N_AX; ++i) m = fmaxf(m, r[i]);
  float e[N_AX], s = 0.f;
  for (int i = 0; i < N_AX; ++i) { e[i] = expf(r[i] - m); s += e[i]; }
  float inv = 1.f / s;
  for (int i = 0; i < N_AX; ++i) px[t * N_AX + i] = e[i] * inv;
}

// --- prep: py softmax over y (axis=1), per (t,g) ---------------------------
__global__ void k_py(const float* __restrict__ pv, float* __restrict__ py) {
  int id = blockIdx.x * blockDim.x + threadIdx.x;   // t*64 + g
  if (id >= LEN_H * N_G) return;
  int t = id >> 6, g = id & 63;
  const float* base = pv + LEN_H * N_AX + (size_t)t * N_AY * N_G + g;
  float m = base[0];
  for (int y = 1; y < N_AY; ++y) m = fmaxf(m, base[y * N_G]);
  float e[N_AY], s = 0.f;
  for (int y = 0; y < N_AY; ++y) { e[y] = expf(base[y * N_G] - m); s += e[y]; }
  float inv = 1.f / s;
  for (int y = 0; y < N_AY; ++y) py[(size_t)t * N_AY * N_G + y * N_G + g] = e[y] * inv;
}

// --- prep: q[t,s,y] = sum_g py[t,y,g] * obv[s,g] ---------------------------
__global__ void k_q(const float* __restrict__ py, const float* __restrict__ obv,
                    float* __restrict__ q) {
  int id = blockIdx.x * blockDim.x + threadIdx.x;   // t*8192 + s*16 + y
  int y = id & 15, s = (id >> 4) & 511, t = id >> 13;
  const float* pyr = py + (size_t)t * 1024 + y * 64;
  const float* ob  = obv + (size_t)s * 64;
  float a = 0.f;
  for (int g = 0; g < N_G; ++g) a += pyr[g] * ob[g];
  q[id] = a;
}

// --- prep: eterm/cterm[t,s] = sum_xy px*q*(E | cost) -----------------------
__global__ void k_ec(const float* __restrict__ E, const float* __restrict__ q,
                     const float* __restrict__ px, const float* __restrict__ cx,
                     const float* __restrict__ cy, float* __restrict__ eterm,
                     float* __restrict__ cterm) {
  int id = blockIdx.x * blockDim.x + threadIdx.x;   // t*512 + s
  int s = id & 511, t = id >> 9;
  float cxt = cx[t], cyt = cy[t];
  const float* qe = q + (size_t)id * 16;
  float ea = 0.f, ca = 0.f;
  for (int x = 0; x < 16; ++x) {
    float wx = px[t * 16 + x];
    for (int y = 0; y < 16; ++y) {
      float w = wx * qe[y];
      ea += w * E[(size_t)s * 256 + x * 16 + y];
      ca += w * (cxt * (float)x + cyt * (float)y);
    }
  }
  eterm[id] = ea; cterm[id] = ca;
}

// --- scan carry init: Vr = E[:,0,0], Vc = 0 --------------------------------
__global__ void k_vinit(const float* __restrict__ E, float* __restrict__ vbuf0) {
  int s = threadIdx.x;                 // 512 threads
  vbuf0[s]       = E[(size_t)s * 256];
  vbuf0[512 + s] = 0.f;
}

// --- per step: bf16 carry + accumulator init with E/cost terms -------------
__global__ void k_step_init(const float* __restrict__ vin, float* __restrict__ vout,
                            unsigned short* __restrict__ vbf,
                            const float* __restrict__ et, const float* __restrict__ ct) {
  int s = threadIdx.x;                 // 512 threads
  vbf[s]        = f32_to_bf16_rne(vin[s]);
  vbf[512 + s]  = f32_to_bf16_rne(vin[512 + s]);
  vout[s]       = et[s];
  vout[512 + s] = ct[s];
}

// --- per step main: WMMA bf16 matvec + weighted reduction ------------------
// One block per state s; wave w handles the 16-row tile (x=w, y=0..15).
// B columns: 0 = Vr, 1 = Vc (rest zero, via zeroed LDS row 2 -> branchless).
__global__ __launch_bounds__(512) void k_step_main(
    const unsigned short* __restrict__ T4bf,   // [131072][512] bf16, row-major
    const unsigned short* __restrict__ vbf,    // [2][512] bf16 (Vr, Vc) global
    const float* __restrict__ q_t,             // [512][16]
    const float* __restrict__ px_t,            // [16]
    float* __restrict__ vout) {                // [1024] pre-seeded with e/c terms
  // Single LDS object => base offset 0 (required by the TDM descriptor below).
  __shared__ struct __align__(16) {
    unsigned short vb[3 * 512];   // rows: 0=Vr, 1=Vc, 2=zeros
    float red_r[16];
    float red_c[16];
  } sm;

  const int tid  = threadIdx.x;
  const int wave = tid >> 5;                   // = x
  const int lane = tid & 31;
  const int l15  = lane & 15;                  // A: row M.  B/D: column N.
  const int half = lane >> 4;

  // ---- stage B table into LDS -------------------------------------------
#if __has_builtin(__builtin_amdgcn_tensor_load_to_lds) && __has_builtin(__builtin_amdgcn_s_wait_tensorcnt)
  if (wave == 0) {
    // Tensor DMA: 1-D tile, 256 x 8B = 2KB, global vbf -> LDS offset 0.
    unsigned long long ga = (unsigned long long)(const void*)vbf;
    v4u g0 = { 1u,                                   // count=1 descriptor
               0u,                                   // lds_addr = 0
               (unsigned)ga,                         // global_addr[31:0]
               ((unsigned)(ga >> 32) & 0x01FFFFFFu) | 0x80000000u }; // addr[56:32] | type=2
    v8i g1 = { 0x00030000,                           // data_size = 3 (8B units)
               0x01000000,                           // tensor_dim0 = 256 (lo16 in [31:16])
               0x00010000,                           // tensor_dim0 hi16=0 | tensor_dim1 = 1
               0x01000000,                           // tile_dim0 = 256 in [31:16]
               1,                                    // tile_dim1 = 1
               256,                                  // tensor_dim0_stride = 256
               0, 0 };
    v4i gz = { 0, 0, 0, 0 };
#if __clang_major__ >= 23
    v8i gz8 = { 0, 0, 0, 0, 0, 0, 0, 0 };
    __builtin_amdgcn_tensor_load_to_lds(g0, g1, gz, gz, gz8, 0);
#else
    __builtin_amdgcn_tensor_load_to_lds(g0, g1, gz, gz, 0);
#endif
    __builtin_amdgcn_s_wait_tensorcnt(0);
  }
  asm volatile("" ::: "memory");   // LDS contents written behind the compiler's back
#else
  if (tid < 128) ((u32x4*)sm.vb)[tid] = ((const u32x4*)vbf)[tid];  // 2KB copy
#endif
  if (tid < 64) {                  // zero row 2 (1KB)
    u32x4 z = {0u, 0u, 0u, 0u};
    ((u32x4*)(sm.vb + 2 * 512))[tid] = z;
  }
  __syncthreads();

  // ---- hot loop ----------------------------------------------------------
  const size_t rowBase = ((size_t)blockIdx.x * 256 + wave * 16 + l15) * 512;
  const int kA  = half * 8;
  const int row = (l15 < 2) ? l15 : 2;         // branchless column select
  const unsigned short* bbase = sm.vb + row * 512 + half * 8;

  union AF { v16bf v; u32x4 u[2]; };
  union BF { v16bf v; u32x4 u[2]; };
  v8f acc0 = {0.f, 0.f, 0.f, 0.f, 0.f, 0.f, 0.f, 0.f};
  v8f acc1 = {0.f, 0.f, 0.f, 0.f, 0.f, 0.f, 0.f, 0.f};

#pragma unroll
  for (int kk = 0; kk < 16; kk += 2) {
    const int k0 = kk * 32;
    if (kk < 8)  // stream this tile's rows ahead through L0 (WGP scope)
      __builtin_prefetch((const void*)(T4bf + rowBase + k0 + 256), 0, 3);
    // 16-bit A 16x32 layout: two contiguous 16B chunks per lane.
    AF a0, a1; BF b0, b1;
    a0.u[0] = *reinterpret_cast<const u32x4*>(T4bf + rowBase + (k0 + kA));
    a0.u[1] = *reinterpret_cast<const u32x4*>(T4bf + rowBase + (k0 + kA + 16));
    b0.u[0] = *reinterpret_cast<const u32x4*>(bbase + k0);
    b0.u[1] = *reinterpret_cast<const u32x4*>(bbase + k0 + 16);
    a1.u[0] = *reinterpret_cast<const u32x4*>(T4bf + rowBase + (k0 + 32 + kA));
    a1.u[1] = *reinterpret_cast<const u32x4*>(T4bf + rowBase + (k0 + 32 + kA + 16));
    b1.u[0] = *reinterpret_cast<const u32x4*>(bbase + k0 + 32);
    b1.u[1] = *reinterpret_cast<const u32x4*>(bbase + k0 + 48);
    acc0 = __builtin_amdgcn_wmma_f32_16x16x32_bf16(false, a0.v, false, b0.v,
                                                   (short)0, acc0, false, false);
    acc1 = __builtin_amdgcn_wmma_f32_16x16x32_bf16(false, a1.v, false, b1.v,
                                                   (short)0, acc1, false, false);
  }
  v8f acc = acc0 + acc1;

  // D: lane holds column l15, rows half*8+v.  Weight rows by q[s,y].
  const float* qrow = q_t + blockIdx.x * 16 + half * 8;
  float p = 0.f;
#pragma unroll
  for (int v = 0; v < 8; ++v) p += acc[v] * qrow[v];
  p += __shfl_xor(p, 16, 32);                  // combine row halves per column
  float w = px_t[wave];
  if (lane == 0) sm.red_r[wave] = w * p;       // column 0 -> Phi_r dot q
  if (lane == 1) sm.red_c[wave] = w * p;       // column 1 -> Phi_c dot q
  __syncthreads();
  if (tid == 0) {
    float a = 0.f;
    for (int i = 0; i < 16; ++i) a += sm.red_r[i];
    vout[blockIdx.x] += a;                     // deterministic: single writer
  } else if (tid == 1) {
    float a = 0.f;
    for (int i = 0; i < 16; ++i) a += sm.red_c[i];
    vout[512 + blockIdx.x] += a;
  }
}

// --- finalize: J_r = pi0.Vr0, J_c = pi0.Vc0 --------------------------------
__global__ void k_final(const float* __restrict__ v0, const float* __restrict__ pi0,
                        const int* __restrict__ lam, float* __restrict__ out) {
  if (threadIdx.x != 0 || blockIdx.x != 0) return;
  float jr = 0.f, jc = 0.f;
  for (int s = 0; s < N_M; ++s) {
    jr += pi0[s] * v0[s];
    jc += pi0[s] * v0[512 + s];
  }
  float l = (float)(*lam);
  out[0] = jr + l * jc;
  out[1] = jr;
  out[2] = jc;
}

extern "C" void kernel_launch(void* const* d_in, const int* in_sizes, int n_in,
                              void* d_out, int out_size, void* d_ws, size_t ws_size,
                              hipStream_t stream) {
  const float* policy = (const float*)d_in[0];
  const float* E      = (const float*)d_in[1];
  const float* T4     = (const float*)d_in[2];
  const float* obv    = (const float*)d_in[3];
  const float* cx     = (const float*)d_in[4];
  const float* cy     = (const float*)d_in[5];
  const float* pi0    = (const float*)d_in[6];
  const int*   lam    = (const int*)d_in[7];
  float* out = (float*)d_out;

  // Carve workspace (~145 MB).
  char* p = (char*)d_ws;
  auto take = [&](size_t bytes) -> char* {
    char* r = p;
    p += (bytes + 255) & ~(size_t)255;
    return r;
  };
  unsigned short* T4bf = (unsigned short*)take(T4_ELEMS * 2);                  // 128 MB
  float* q     = (float*)take((size_t)LEN_H * N_M * N_AY * 4);                 // 8 MB
  float* py    = (float*)take((size_t)LEN_H * N_AY * N_G * 4);                 // 1 MB
  float* px    = (float*)take((size_t)LEN_H * N_AX * 4);
  float* eterm = (float*)take((size_t)LEN_H * N_M * 4);
  float* cterm = (float*)take((size_t)LEN_H * N_M * 4);
  float* vbuf  = (float*)take(2 * 1024 * 4);           // ping-pong fp32 carries
  unsigned short* vbf = (unsigned short*)take(1024 * 2);
  if ((size_t)(p - (char*)d_ws) > ws_size) return;     // workspace too small

  // One-time prep (all tiny except the 384MB-traffic convert).
  k_convert_t4<<<4096, 256, 0, stream>>>(T4, T4bf);
  k_px<<<1, 256, 0, stream>>>(policy, px);
  k_py<<<(LEN_H * N_G + 255) / 256, 256, 0, stream>>>(policy, py);
  k_q<<<(LEN_H * N_M * N_AY) / 256, 256, 0, stream>>>(py, obv, q);
  k_ec<<<(LEN_H * N_M) / 256, 256, 0, stream>>>(E, q, px, cx, cy, eterm, cterm);
  k_vinit<<<1, 512, 0, stream>>>(E, vbuf);             // parity of carry-in at t=255 is 0

  // Reverse scan: carry-in = vbuf[(t+1)&1], carry-out = vbuf[t&1].
  for (int t = LEN_H - 1; t >= 0; --t) {
    const float* vin  = vbuf + ((t + 1) & 1) * 1024;
    float*       vout = vbuf + (t & 1) * 1024;
    k_step_init<<<1, 512, 0, stream>>>(vin, vout, vbf, eterm + (size_t)t * N_M,
                                       cterm + (size_t)t * N_M);
    k_step_main<<<N_M, 512, 0, stream>>>(T4bf, vbf, q + (size_t)t * N_M * N_AY,
                                         px + (size_t)t * N_AX, vout);
  }
  // t=0 wrote parity 0.
  k_final<<<1, 32, 0, stream>>>(vbuf, pi0, lam, out);
}
